// RecurrentLayer_34479997452834
// MI455X (gfx1250) — matrix operand order repaired and checked by reading, not compiled, not run on hardware
//
#include <hip/hip_runtime.h>
#include <hip/hip_bf16.h>

// GRU recurrent layer, MI455X (gfx1250) bf16-WMMA pipeline with async
// global->LDS staging (ASYNCcnt) and double-buffered LDS tiles.
//
// Phases:
//  P0: convert X -> bf16; transpose+convert W_ih/W_hh/W_out -> bf16 [N,K]; h0 -> bf16
//  P1: gi[B*T,3H] = Xb @ W_ihT          (parallel WMMA GEMM, bf16 out)
//  P2: for t in 0..T-1: fused step kernel
//        gh = h_t @ W_hhT (3 gate accumulations), GRU gate math, h_{t+1} -> Hall[t+1]
//  P3: out[B,T,O] = Hall[1..T] @ W_outT + b_out  (parallel WMMA GEMM, f32 out)

#define BDIM 64
#define TDIM 1024
#define IDIM 512
#define HDIM 1024
#define ODIM 256
#define GDIM (3 * HDIM)

typedef __bf16 bf16;
typedef __attribute__((ext_vector_type(8)))  bf16  v8bf;
typedef __attribute__((ext_vector_type(16))) bf16  v16bf;
typedef __attribute__((ext_vector_type(8)))  float v8f;

static_assert(sizeof(bf16) == 2, "bf16 must be 2 bytes");

__device__ __forceinline__ unsigned short f2bf(float f) {
  union { float f; unsigned u; } v; v.f = f;
  unsigned u = v.u;
  u += 0x7fffu + ((u >> 16) & 1u);   // round-to-nearest-even
  return (unsigned short)(u >> 16);
}
__device__ __forceinline__ float bf2f(unsigned short s) {
  union { unsigned u; float f; } v; v.u = ((unsigned)s) << 16;
  return v.f;
}
__device__ __forceinline__ v16bf cat8(v8bf lo, v8bf hi) {
  return __builtin_shufflevector(lo, hi, 0, 1, 2, 3, 4, 5, 6, 7,
                                 8, 9, 10, 11, 12, 13, 14, 15);
}

// ---- CDNA5 async global->LDS copy (16B per lane), tracked by ASYNCcnt ----
// GV addressing: 64-bit global VA in a VGPR pair; LDS dest byte offset =
// low 32 bits of the generic shared-aperture address (aperture truncation).
__device__ __forceinline__ void async_load16(const unsigned short* g,
                                             unsigned short* l) {
  unsigned lds_off = (unsigned)(unsigned long long)l;
  unsigned long long ga = (unsigned long long)g;
  asm volatile("global_load_async_to_lds_b128 %0, %1, off"
               :: "v"(lds_off), "v"(ga)
               : "memory");
}

template <int N>
__device__ __forceinline__ void wait_async() {
  asm volatile("s_wait_asynccnt %0" :: "n"(N) : "memory");
}

// ---------------- P0: conversions ----------------

__global__ __launch_bounds__(256)
void cvt_f32_bf16(const float* __restrict__ in, unsigned short* __restrict__ out,
                  unsigned n) {
  unsigned i = (blockIdx.x * 256u + threadIdx.x) * 4u;
  if (i >= n) return;
  float4 f = *(const float4*)(in + i);
  ushort4 o;
  o.x = f2bf(f.x); o.y = f2bf(f.y); o.z = f2bf(f.z); o.w = f2bf(f.w);
  *(ushort4*)(out + i) = o;
}

// W[K,N] f32 row-major -> WT[N,K] bf16 row-major
__global__ __launch_bounds__(256)
void cvt_transpose(const float* __restrict__ W, unsigned short* __restrict__ WT,
                   int K, int N) {
  unsigned idx = blockIdx.x * 256u + threadIdx.x;
  if (idx >= (unsigned)(K * N)) return;
  unsigned n = idx / (unsigned)K;
  unsigned k = idx % (unsigned)K;
  WT[(size_t)n * K + k] = f2bf(W[(size_t)k * N + n]);
}

// ---------------- P1: bf16 GEMM, C[M,N](bf16) = A[M,K] @ BT[N,K] ----------------
// block = 256 threads (8 wave32), tile 128M x 64N, K-step 32,
// double-buffered async LDS staging (3 async b128 per thread per tile).

__global__ __launch_bounds__(256)
void gemm_bf16(const unsigned short* __restrict__ A,
               const unsigned short* __restrict__ BT,
               unsigned short* __restrict__ C,
               int M, int N, int K) {
  __shared__ unsigned short As[2][128 * 32];
  __shared__ unsigned short Bs[2][64 * 32];
  const int tid = threadIdx.x;
  const int wave = tid >> 5, lane = tid & 31;
  const int lh = lane >> 4, ll = lane & 15;
  const int m0 = blockIdx.x * 128;
  const int n0 = blockIdx.y * 64;
  const int mtb = (wave & 3) * 2;   // 2 M-subtiles per wave
  const int ntb = (wave >> 2) * 2;  // 2 N-subtiles per wave
  const int srow = tid >> 2, sseg = tid & 3;

  v8f acc[2][2] = {};
  const int NT = K >> 5;

  // stage tile kt into buffer buf: A 128x32 (512 segs), B 64x32 (256 segs)
  auto stage = [&](int k0, int buf) {
#pragma unroll
    for (int r = 0; r < 2; ++r) {
      int idx = tid + r * 256;
      int row = idx >> 2, seg = idx & 3;
      async_load16(A + (size_t)(m0 + row) * K + k0 + seg * 8,
                   &As[buf][row * 32 + seg * 8]);
    }
    async_load16(BT + (size_t)(n0 + srow) * K + k0 + sseg * 8,
                 &Bs[buf][srow * 32 + sseg * 8]);
  };

  stage(0, 0);
  for (int kt = 0; kt < NT; ++kt) {
    const int buf = kt & 1;
    if (kt + 1 < NT) { stage((kt + 1) << 5, buf ^ 1); wait_async<3>(); }
    else             { wait_async<0>(); }
    __syncthreads();

    v16bf a[2], b[2];
#pragma unroll
    for (int i = 0; i < 2; ++i) {
      // A frag (16x32 bf16): lanes 0-15 K{0..7,16..23}, lanes 16-31 K{8..15,24..31}
      const bf16* ab = (const bf16*)&As[buf][(mtb + i) * 16 * 32];
      a[i] = cat8(*(const v8bf*)(ab + ll * 32 + lh * 8),
                  *(const v8bf*)(ab + ll * 32 + 16 + lh * 8));
      // B frag (32x16 bf16): lanes 0-15 K{0..7,16..23}, lanes 16-31 K{8..15,24..31}
      const bf16* bb = (const bf16*)&Bs[buf][(ntb + i) * 16 * 32];
      b[i] = cat8(*(const v8bf*)(bb + ll * 32 + lh * 8),
                  *(const v8bf*)(bb + ll * 32 + 16 + lh * 8));
    }
#pragma unroll
    for (int i = 0; i < 2; ++i)
#pragma unroll
      for (int j = 0; j < 2; ++j)
        acc[i][j] = __builtin_amdgcn_wmma_f32_16x16x32_bf16(
            false, a[i], false, b[j], (short)0, acc[i][j], false, false);
    __syncthreads();
  }

#pragma unroll
  for (int i = 0; i < 2; ++i)
#pragma unroll
    for (int j = 0; j < 2; ++j) {
      int col = n0 + (ntb + j) * 16 + ll;
#pragma unroll
      for (int v = 0; v < 8; ++v) {
        int row = m0 + (mtb + i) * 16 + lh * 8 + v;  // C/D layout: M = v + 8*half
        C[(size_t)row * N + col] = f2bf(acc[i][j][v]);
      }
    }
}

// ---------------- P2: fused GRU step ----------------
// grid = H/64 blocks; each block: columns j0..j0+63 of h_new, all 64 batch rows,
// three gate GEMM accumulations (r,z,n) fused with nonlinearity + state update.
// Double-buffered async staging: 4 async b128 per thread per K-tile
// (Hs 64x32 = 256 segs, Ws 3x64x32 = 768 segs -> 1024 segs / 256 thr).

__global__ __launch_bounds__(256)
void gru_step(unsigned short* __restrict__ Hall,       // [(T+1),B,H] bf16
              const unsigned short* __restrict__ WhhT, // [3H,H] bf16
              const unsigned short* __restrict__ gi,   // [B,T,3H] bf16
              const float* __restrict__ b_ih,
              const float* __restrict__ b_hh,
              int t) {
  __shared__ unsigned short Hs[2][64 * 32];       // h tile   [b][k]
  __shared__ unsigned short Ws[2][3 * 64 * 32];   // W tiles  [g][col][k]
  const unsigned short* hprev = Hall + (size_t)t * (BDIM * HDIM);
  unsigned short* hnext = Hall + (size_t)(t + 1) * (BDIM * HDIM);
  const int tid = threadIdx.x;
  const int wave = tid >> 5, lane = tid & 31;
  const int lh = lane >> 4, ll = lane & 15;
  const int j0 = blockIdx.x * 64;
  const int mt = wave & 3;   // M-subtile (batch rows mt*16..)
  const int np = wave >> 2;  // N-pair (cols np*32..np*32+31)
  const int srow = tid >> 2, sseg = tid & 3;

  auto stage = [&](int k0, int buf) {
    async_load16(hprev + (size_t)srow * HDIM + k0 + sseg * 8,
                 &Hs[buf][srow * 32 + sseg * 8]);
#pragma unroll
    for (int g = 0; g < 3; ++g)
      async_load16(WhhT + (size_t)(g * HDIM + j0 + srow) * HDIM + k0 + sseg * 8,
                   &Ws[buf][(g * 64 + srow) * 32 + sseg * 8]);
  };

  v8f acc[3][2] = {};
  constexpr int NT = HDIM / 32;
  stage(0, 0);
  for (int kt = 0; kt < NT; ++kt) {
    const int buf = kt & 1;
    if (kt + 1 < NT) { stage((kt + 1) * 32, buf ^ 1); wait_async<4>(); }
    else             { wait_async<0>(); }
    __syncthreads();

    const bf16* hb = (const bf16*)&Hs[buf][(mt * 16) * 32];
    v16bf a = cat8(*(const v8bf*)(hb + ll * 32 + lh * 8),
                   *(const v8bf*)(hb + ll * 32 + 16 + lh * 8));
#pragma unroll
    for (int g = 0; g < 3; ++g)
#pragma unroll
      for (int ni = 0; ni < 2; ++ni) {
        const bf16* wb = (const bf16*)&Ws[buf][(g * 64 + (np * 2 + ni) * 16) * 32];
        v16bf b = cat8(*(const v8bf*)(wb + ll * 32 + lh * 8),
                       *(const v8bf*)(wb + ll * 32 + 16 + lh * 8));
        acc[g][ni] = __builtin_amdgcn_wmma_f32_16x16x32_bf16(
            false, a, false, b, (short)0, acc[g][ni], false, false);
      }
    __syncthreads();
  }

  // Gate math + state update.  Lane owns column j, rows m = mt*16 + 8*lh + v.
#pragma unroll
  for (int ni = 0; ni < 2; ++ni) {
    const int j = j0 + (np * 2 + ni) * 16 + ll;
    const float bir = b_ih[j],            bhr = b_hh[j];
    const float biz = b_ih[HDIM + j],     bhz = b_hh[HDIM + j];
    const float bin = b_ih[2 * HDIM + j], bhn = b_hh[2 * HDIM + j];
#pragma unroll
    for (int v = 0; v < 8; ++v) {
      const int m = mt * 16 + lh * 8 + v;  // batch index
      const size_t gb = ((size_t)m * TDIM + t) * GDIM;
      const float ir = bf2f(gi[gb + j]) + bir;
      const float iz = bf2f(gi[gb + HDIM + j]) + biz;
      const float in_ = bf2f(gi[gb + 2 * HDIM + j]) + bin;
      const float hr = acc[0][ni][v] + bhr;
      const float hz = acc[1][ni][v] + bhz;
      const float hn = acc[2][ni][v] + bhn;
      const float r = 1.f / (1.f + __expf(-(ir + hr)));
      const float z = 1.f / (1.f + __expf(-(iz + hz)));
      const float n = tanhf(in_ + r * hn);
      const float hp = bf2f(hprev[(size_t)m * HDIM + j]);
      hnext[(size_t)m * HDIM + j] = f2bf((1.f - z) * n + z * hp);
    }
  }
}

// ---------------- P3: output projection, f32 out with bias + [t*B+b] -> [b,t] permute ----

__global__ __launch_bounds__(256)
void outproj(const unsigned short* __restrict__ Hseq,   // [T*B, H] bf16 (= Hall+B*H)
             const unsigned short* __restrict__ WoutT,  // [O, H] bf16
             const float* __restrict__ b_out,
             float* __restrict__ out) {                 // [B, T, O] f32
  __shared__ unsigned short As[2][128 * 32];
  __shared__ unsigned short Bs[2][64 * 32];
  const int tid = threadIdx.x;
  const int wave = tid >> 5, lane = tid & 31;
  const int lh = lane >> 4, ll = lane & 15;
  const int m0 = blockIdx.x * 128;
  const int n0 = blockIdx.y * 64;
  const int mtb = (wave & 3) * 2;
  const int ntb = (wave >> 2) * 2;
  const int srow = tid >> 2, sseg = tid & 3;

  auto stage = [&](int k0, int buf) {
#pragma unroll
    for (int r = 0; r < 2; ++r) {
      int idx = tid + r * 256;
      int row = idx >> 2, seg = idx & 3;
      async_load16(Hseq + (size_t)(m0 + row) * HDIM + k0 + seg * 8,
                   &As[buf][row * 32 + seg * 8]);
    }
    async_load16(WoutT + (size_t)(n0 + srow) * HDIM + k0 + sseg * 8,
                 &Bs[buf][srow * 32 + sseg * 8]);
  };

  v8f acc[2][2] = {};
  constexpr int NT = HDIM / 32;
  stage(0, 0);
  for (int kt = 0; kt < NT; ++kt) {
    const int buf = kt & 1;
    if (kt + 1 < NT) { stage((kt + 1) * 32, buf ^ 1); wait_async<3>(); }
    else             { wait_async<0>(); }
    __syncthreads();

    v16bf a[2], b[2];
#pragma unroll
    for (int i = 0; i < 2; ++i) {
      const bf16* ab = (const bf16*)&As[buf][(mtb + i) * 16 * 32];
      a[i] = cat8(*(const v8bf*)(ab + ll * 32 + lh * 8),
                  *(const v8bf*)(ab + ll * 32 + 16 + lh * 8));
      const bf16* bb = (const bf16*)&Bs[buf][(ntb + i) * 16 * 32];
      b[i] = cat8(*(const v8bf*)(bb + ll * 32 + lh * 8),
                  *(const v8bf*)(bb + ll * 32 + 16 + lh * 8));
    }
#pragma unroll
    for (int i = 0; i < 2; ++i)
#pragma unroll
      for (int j = 0; j < 2; ++j)
        acc[i][j] = __builtin_amdgcn_wmma_f32_16x16x32_bf16(
            false, a[i], false, b[j], (short)0, acc[i][j], false, false);
    __syncthreads();
  }

#pragma unroll
  for (int i = 0; i < 2; ++i)
#pragma unroll
    for (int j = 0; j < 2; ++j) {
      int col = n0 + (ntb + j) * 16 + ll;
      float bo = b_out[col];
#pragma unroll
      for (int v = 0; v < 8; ++v) {
        int row = m0 + (mtb + i) * 16 + lh * 8 + v;  // row = t*B + b
        int tt = row >> 6, bb2 = row & 63;           // B == 64
        out[((size_t)bb2 * TDIM + tt) * ODIM + col] = acc[i][j][v] + bo;
      }
    }
}

// ---------------- launch ----------------

extern "C" void kernel_launch(void* const* d_in, const int* in_sizes, int n_in,
                              void* d_out, int out_size, void* d_ws, size_t ws_size,
                              hipStream_t stream) {
  (void)in_sizes; (void)n_in; (void)out_size; (void)ws_size;
  const float* X    = (const float*)d_in[0];
  const float* h0   = (const float*)d_in[1];
  const float* Wih  = (const float*)d_in[2];
  const float* Whh  = (const float*)d_in[3];
  const float* bih  = (const float*)d_in[4];
  const float* bhh  = (const float*)d_in[5];
  const float* Wout = (const float*)d_in[6];
  const float* bout = (const float*)d_in[7];
  float* out = (float*)d_out;

  char* ws = (char*)d_ws;
  size_t off = 0;
  auto take = [&](size_t bytes) -> void* {
    void* p = ws + off;
    off += (bytes + 255) & ~(size_t)255;
    return p;
  };
  unsigned short* Xb    = (unsigned short*)take((size_t)BDIM * TDIM * IDIM * 2); //  64 MB
  unsigned short* WihT  = (unsigned short*)take((size_t)GDIM * IDIM * 2);        //   3 MB
  unsigned short* WhhT  = (unsigned short*)take((size_t)GDIM * HDIM * 2);        //   6 MB
  unsigned short* WoutT = (unsigned short*)take((size_t)ODIM * HDIM * 2);        // 0.5 MB
  unsigned short* gi    = (unsigned short*)take((size_t)BDIM * TDIM * GDIM * 2); // 384 MB
  unsigned short* Hall  = (unsigned short*)take((size_t)(TDIM + 1) * BDIM * HDIM * 2); // 128 MB

  // P0: conversions (all sizes are multiples of 4/256)
  {
    unsigned n = (unsigned)BDIM * TDIM * IDIM;
    cvt_f32_bf16<<<dim3(n / 1024), dim3(256), 0, stream>>>(X, Xb, n);
  }
  {
    unsigned n = (unsigned)BDIM * HDIM;  // h0 -> Hall slot 0
    cvt_f32_bf16<<<dim3(n / 1024), dim3(256), 0, stream>>>(h0, Hall, n);
  }
  cvt_transpose<<<dim3((IDIM * GDIM + 255) / 256), dim3(256), 0, stream>>>(Wih, WihT, IDIM, GDIM);
  cvt_transpose<<<dim3((HDIM * GDIM + 255) / 256), dim3(256), 0, stream>>>(Whh, WhhT, HDIM, GDIM);
  cvt_transpose<<<dim3((HDIM * ODIM + 255) / 256), dim3(256), 0, stream>>>(Wout, WoutT, HDIM, ODIM);

  // P1: gi = Xb @ WihT   (65536 x 3072 x 512)
  gemm_bf16<<<dim3((BDIM * TDIM) / 128, GDIM / 64), dim3(256), 0, stream>>>(
      Xb, WihT, gi, BDIM * TDIM, GDIM, IDIM);

  // P2: sequential recurrence, graph-captured launches
  for (int t = 0; t < TDIM; ++t)
    gru_step<<<dim3(HDIM / 64), dim3(256), 0, stream>>>(Hall, WhhT, gi, bih, bhh, t);

  // P3: out = Hall[1..T] @ WoutT + b_out  (65536 x 256 x 1024), permuted store
  outproj<<<dim3((TDIM * BDIM) / 128, ODIM / 64), dim3(256), 0, stream>>>(
      Hall + (size_t)BDIM * HDIM, WoutT, bout, out);
}
